// HDGC_26147760898240
// MI455X (gfx1250) — matrix operand order repaired and checked by reading, not compiled, not run on hardware
//
#include <hip/hip_runtime.h>
#include <cstdint>

// ---------------------------------------------------------------------------
// HD-GC block for MI455X (gfx1250, wave32, WMMA bf16 16x16x32, f32 accum).
// Shapes: N=64 C=256 T=64 V=25 H=3 D=32 O=256, B = N*T = 4096 frames.
// CDNA5 paths used: v_wmma_f32_16x16x32_bf16, tensor_load_to_lds (TDM) +
// s_wait_tensorcnt, global_load_async_to_lds_b128 + s_wait_asynccnt,
// global_prefetch_b8.
// Workspace layout (bytes), total ~137 MB:
//   featbf : B*32*256 bf16          @ 0          (64 MB)
//   pp_bf  : B*32*192 bf16          @ 67108864   (48 MB)  (phi | psi)
//   Af_bf  : B*3*32*32 bf16         @ 117440512  (24 MB)
//   Wpp_bf : 192*256 bf16           @ 142606336
//   Wd_bf  : 3*256*256 bf16         @ 142704640
//   Wg_bf  : 256*256 bf16           @ 143097856
// ---------------------------------------------------------------------------

typedef __bf16 bf16_t;
typedef bf16_t   v16bf __attribute__((ext_vector_type(16)));
typedef float    v8f   __attribute__((ext_vector_type(8)));
typedef uint32_t u32x4 __attribute__((ext_vector_type(4)));
typedef uint32_t u32x8 __attribute__((ext_vector_type(8)));

#define NN 64
#define CC 256
#define TT 64
#define VV 25
#define VP 32
#define HH 3
#define DD 32
#define OO 256
#define BB (NN * TT)
#define EE 192   // H*D*2 (phi + psi channels)

// ---- WMMA fragment loaders (contiguous k = 16*(lane>>4) + e, same order for A and B, wave32) ----

// A matrix 16x32 bf16, source stored row-major [M][K], leading dim ld.
// lane: m = lane&15, hi = lane>>4; elem j -> K = hi*16+j
// => one 32B contiguous read per lane.
__device__ __forceinline__ v16bf load_A_MK(const bf16_t* base, int ld,
                                           int m_off, int k_off) {
  int lane = threadIdx.x & 31;
  const bf16_t* row = base + (size_t)(m_off + (lane & 15)) * ld + k_off
                      + (lane >> 4) * 16;
  union { uint4 q[2]; v16bf v; } u;
  u.q[0] = *(const uint4*)(row);
  u.q[1] = *(const uint4*)(row + 8);
  return u.v;
}

// B matrix 32x16 bf16, operand stored TRANSPOSED [N][K] row-major
// (weight layout W[o][c], k=c). lane: n = lane&15, elem e -> K=(lane>>4)*16+e
// => one 32B contiguous read per lane.
__device__ __forceinline__ v16bf load_B_NK(const bf16_t* base, int ld,
                                           int n_off, int k_off) {
  int lane = threadIdx.x & 31;
  const bf16_t* row = base + (size_t)(n_off + (lane & 15)) * ld + k_off
                      + (lane >> 4) * 16;
  union { uint4 q[2]; v16bf v; } u;
  u.q[0] = *(const uint4*)(row);
  u.q[1] = *(const uint4*)(row + 8);
  return u.v;
}

// B matrix 32x16 bf16, operand stored natural [K][N] row-major (feat as B in
// the z GEMM). Column-strided per-lane reads (LDS source).
__device__ __forceinline__ v16bf load_B_KN(const bf16_t* base, int ld,
                                           int n_off, int k_off) {
  int lane = threadIdx.x & 31;
  int n    = n_off + (lane & 15);
  int kb   = k_off + (lane >> 4) * 16;
  v16bf b;
#pragma unroll
  for (int e = 0; e < 16; ++e) b[e] = base[(size_t)(kb + e) * ld + n];
  return b;
}

__device__ __forceinline__ v8f wmma_bf16(v16bf a, v16bf b, v8f c) {
  return __builtin_amdgcn_wmma_f32_16x16x32_bf16(
      /*neg_a=*/false, a, /*neg_b=*/false, b,
      /*c_mod=*/(short)0, c, /*reuse_a=*/false, /*reuse_b=*/false);
}

// ---- CDNA5 async data movers -----------------------------------------------

// Tensor Data Mover: DMA a contiguous 16 KB tile (2048 x 8B) global -> LDS.
// D# per CDNA5 ISA: group0 {count=1, lds_addr,
// global_addr[56:0], type=2}; group1 {data_size=3 (8B), tensor_dim0=2048,
// tensor_dim1=1, tile_dim0=2048, tile_dim1=1, stride0=2048}.
__device__ __forceinline__ void tdm_load_16k(uint32_t lds_addr, const void* g) {
  uint64_t ga = (uint64_t)(uintptr_t)g;
  u32x4 g0;
  g0[0] = 1u;                                               // count=1
  g0[1] = lds_addr;                                         // lds_addr
  g0[2] = (uint32_t)ga;                                     // global_addr lo
  g0[3] = (uint32_t)((ga >> 32) & 0x01FFFFFFu) | (2u << 30);// addr[56:32]|type=2
  u32x8 g1;
  g1[0] = (3u << 16);        // workgroup_mask=0, data_size=3 (8 bytes)
  g1[1] = (2048u << 16);     // tensor_dim0[15:0] in bits 63:48
  g1[2] = (1u << 16);        // tensor_dim0[31:16]=0 | tensor_dim1[15:0]=1
  g1[3] = (2048u << 16);     // tensor_dim1[31:16]=0 | tile_dim0=2048
  g1[4] = 1u;                // tile_dim1=1, tile_dim2=0
  g1[5] = 2048u;             // tensor_dim0_stride lo
  g1[6] = 0u;                // stride0 hi, stride1 lo
  g1[7] = 0u;                // stride1 hi
  asm volatile("tensor_load_to_lds %0, %1" :: "s"(g0), "s"(g1) : "memory");
}

__device__ __forceinline__ void wait_tensorcnt0() {
#if __has_builtin(__builtin_amdgcn_s_wait_tensorcnt)
  __builtin_amdgcn_s_wait_tensorcnt((short)0);
#else
  asm volatile("s_wait_tensorcnt 0x0" ::: "memory");
#endif
}

// Async global -> LDS, 16 bytes per active lane (ASYNCcnt-tracked).
__device__ __forceinline__ void async_ld_b128(uint32_t lds_addr, const void* g) {
  asm volatile("global_load_async_to_lds_b128 %0, %1, off"
               :: "v"(lds_addr), "v"((uint64_t)(uintptr_t)g) : "memory");
}

__device__ __forceinline__ void wait_asynccnt0() {
  asm volatile("s_wait_asynccnt 0x0" ::: "memory");
}

// ---------------------------------------------------------------------------
// Kernel 0a: transpose x [N][C][T][V] f32 -> featbf [B][32][256] bf16,
// zero-padding joint rows 25..31 (makes all downstream padding inert).
// ---------------------------------------------------------------------------
__global__ __launch_bounds__(256) void k_pack_feat(const float* __restrict__ x,
                                                   bf16_t* __restrict__ featbf) {
  int b = blockIdx.x;          // b = n*T + t
  int n = b >> 6;
  int t = b & (TT - 1);
  int c = threadIdx.x;         // 0..255
  const float* xc = x + (((size_t)n * CC + c) * TT + t) * VV;
  bf16_t* dst = featbf + (size_t)b * VP * CC;
#pragma unroll 4
  for (int w = 0; w < VP; ++w) {
    float val = (w < VV) ? xc[w] : 0.0f;
    dst[(size_t)w * CC + c] = (bf16_t)val;
  }
}

// ---------------------------------------------------------------------------
// Kernel 0b: weight conversion to bf16: Wpp = [W_phi ; W_psi] (192x256),
// Wd (3x256x256), Wg (256x256).
// ---------------------------------------------------------------------------
#define NW_PP (EE * CC)            // 49152
#define NW_D  (HH * OO * CC)       // 196608
#define NW_G  (OO * CC)            // 65536
__global__ __launch_bounds__(256) void k_pack_w(const float* __restrict__ Wphi,
                                                const float* __restrict__ Wpsi,
                                                const float* __restrict__ Wd,
                                                const float* __restrict__ Wg,
                                                bf16_t* __restrict__ Wpp_bf,
                                                bf16_t* __restrict__ Wd_bf,
                                                bf16_t* __restrict__ Wg_bf) {
  int idx = blockIdx.x * 256 + threadIdx.x;
  if (idx < NW_PP) {
    int e = idx / CC, c = idx % CC;
    float v = (e < 96) ? Wphi[e * CC + c] : Wpsi[(e - 96) * CC + c];
    Wpp_bf[idx] = (bf16_t)v;
  } else if (idx < NW_PP + NW_D) {
    int j = idx - NW_PP;
    Wd_bf[j] = (bf16_t)Wd[j];
  } else if (idx < NW_PP + NW_D + NW_G) {
    int j = idx - NW_PP - NW_D;
    Wg_bf[j] = (bf16_t)Wg[j];
  }
}

// ---------------------------------------------------------------------------
// Kernel 1: pp[b] = feat[b] (32x256) @ Wpp^T (256x192) + bias  -> bf16.
// One block (8 waves) per frame; feat staged via TDM. 24 output tiles
// (2 mt x 12 nt), 3 per wave, 8 k-steps each -> 192 WMMAs / block.
// ---------------------------------------------------------------------------
__global__ __launch_bounds__(256) void k_phipsi(const bf16_t* __restrict__ featbf,
                                                const bf16_t* __restrict__ Wpp,
                                                const float* __restrict__ b_phi,
                                                const float* __restrict__ b_psi,
                                                bf16_t* __restrict__ pp) {
  int b = blockIdx.x;
  __shared__ alignas(16) bf16_t sfeat[VP * CC];   // 16 KB
  if (threadIdx.x < 32) {                          // wave 0 issues the DMA
    tdm_load_16k((uint32_t)(uintptr_t)sfeat, featbf + (size_t)b * VP * CC);
    wait_tensorcnt0();
  }
  // warm the weight slab (L2-resident across 4096 blocks)
  __builtin_prefetch((const char*)Wpp + threadIdx.x * 384, 0, 1);
  __syncthreads();

  int wave = threadIdx.x >> 5;
  int lane = threadIdx.x & 31;
  int hi   = lane >> 4;
  bf16_t* outb = pp + (size_t)b * VP * EE;

  for (int rep = 0; rep < 3; ++rep) {
    int ti = wave + rep * 8;     // 0..23
    int mt = ti & 1;
    int nt = ti >> 1;            // 0..11
    v8f acc = {};
#pragma unroll
    for (int kk = 0; kk < 8; ++kk) {
      v16bf a  = load_A_MK(sfeat, CC, mt * 16, kk * 32);
      v16bf bm = load_B_NK(Wpp, CC, nt * 16, kk * 32);
      acc = wmma_bf16(a, bm, acc);
    }
    int e = nt * 16 + (lane & 15);
    float bias = (e < 96) ? b_phi[e] : b_psi[e - 96];
#pragma unroll
    for (int r = 0; r < 8; ++r) {
      int v = mt * 16 + hi * 8 + r;
      outb[(size_t)v * EE + e] = (bf16_t)(acc[r] + bias);
    }
  }
}

// ---------------------------------------------------------------------------
// Kernel 2: per frame: S_h = phi_h @ psi_h^T (32x32 via 4 WMMAs, K=32),
// masked softmax over w<25, blend with priors -> Af bf16 [b][h][32][32]
// (rows v>=25 and cols w>=25 forced to zero).
// ---------------------------------------------------------------------------
__global__ __launch_bounds__(128) void k_attn(const bf16_t* __restrict__ pp,
                                              const float* __restrict__ A_prior,
                                              const float* __restrict__ A_2hop,
                                              const float* __restrict__ beta_p,
                                              const float* __restrict__ lam_p,
                                              bf16_t* __restrict__ Af) {
  int b = blockIdx.x;
  __shared__ float S[HH * VP * VP];               // 12 KB
  int wave = threadIdx.x >> 5;
  int lane = threadIdx.x & 31;
  const bf16_t* ppb = pp + (size_t)b * VP * EE;

  if (wave < HH) {
    int h = wave;
#pragma unroll
    for (int mt = 0; mt < 2; ++mt)
#pragma unroll
      for (int nt = 0; nt < 2; ++nt) {
        v8f acc = {};
        v16bf a  = load_A_MK(ppb + h * DD, EE, mt * 16, 0);        // phi_h
        v16bf bm = load_B_NK(ppb + 96 + h * DD, EE, nt * 16, 0);   // psi_h^T
        acc = wmma_bf16(a, bm, acc);
        int hi = lane >> 4;
        int nc = nt * 16 + (lane & 15);
#pragma unroll
        for (int r = 0; r < 8; ++r) {
          int m = mt * 16 + hi * 8 + r;
          S[(h * VP + m) * VP + nc] = acc[r];
        }
      }
  }
  __syncthreads();

  float beta = beta_p[0];
  float lam  = fminf(fmaxf(lam_p[0], 0.0f), 1.0f);
  const float scale = 0.17677669529663687f;   // D^-0.5, D=32

  int tid = threadIdx.x;
  if (tid < HH * VP) {
    int h = tid / VP, v = tid % VP;
    bf16_t* out = Af + (((size_t)b * HH + h) * VP + v) * VP;
    if (v < VV) {
      float row[VV];
      float mx = -1e30f;
#pragma unroll
      for (int w = 0; w < VV; ++w) {
        row[w] = S[(h * VP + v) * VP + w] * scale;
        mx = fmaxf(mx, row[w]);
      }
      float sum = 0.0f;
#pragma unroll
      for (int w = 0; w < VV; ++w) { row[w] = __expf(row[w] - mx); sum += row[w]; }
      float inv = 1.0f / sum;
      const float* pr = A_prior + (h * VV + v) * VV;
      const float* p2 = A_2hop  + (h * VV + v) * VV;
#pragma unroll
      for (int w = 0; w < VV; ++w)
        out[w] = (bf16_t)(pr[w] + beta * p2[w] + lam * row[w] * inv);
#pragma unroll
      for (int w = VV; w < VP; ++w) out[w] = (bf16_t)0.0f;
    } else {
#pragma unroll
      for (int w = 0; w < VP; ++w) out[w] = (bf16_t)0.0f;
    }
  }
}

// ---------------------------------------------------------------------------
// Kernel 3: fused z-GEMM + output projection + gate + BN + residual + ReLU.
// feat staged via TDM; Af staged via async-to-LDS. Per head: z_h = Af_h @ feat
// through a 16 KB LDS buffer, then acc += z_h @ Wd[h]^T (24 WMMAs per output
// tile over 3 heads); gate = feat @ Wg^T (8 WMMAs/tile). Epilogue applies BN,
// sigmoid gate, residual x and ReLU, writing [N][O][T][V] f32.
// ---------------------------------------------------------------------------
__global__ __launch_bounds__(256) void k_out(const bf16_t* __restrict__ featbf,
                                             const bf16_t* __restrict__ Af,
                                             const bf16_t* __restrict__ Wd,
                                             const bf16_t* __restrict__ Wg,
                                             const float* __restrict__ b_d,
                                             const float* __restrict__ b_g,
                                             const float* __restrict__ gamma,
                                             const float* __restrict__ betabn,
                                             const float* __restrict__ mean,
                                             const float* __restrict__ var,
                                             const float* __restrict__ x,
                                             float* __restrict__ out) {
  int b = blockIdx.x;
  int n = b >> 6;
  int t = b & (TT - 1);

  __shared__ alignas(16) bf16_t sfeat[VP * CC];       // 16 KB
  __shared__ alignas(16) bf16_t sAf[HH * VP * VP];    //  6 KB
  __shared__ alignas(16) bf16_t sz[VP * CC];          // 16 KB (reused per head)

  if (threadIdx.x < 32) {                              // wave 0: TDM for feat
    tdm_load_16k((uint32_t)(uintptr_t)sfeat, featbf + (size_t)b * VP * CC);
  }
  {                                                    // all waves: async Af
    const char* gsrc = (const char*)(Af + (size_t)b * HH * VP * VP);
    uint32_t lds0 = (uint32_t)(uintptr_t)sAf;
    int i = threadIdx.x;                               // 384 x 16B total
    async_ld_b128(lds0 + i * 16, gsrc + i * 16);
    if (threadIdx.x < 128) {
      int j = 256 + threadIdx.x;
      async_ld_b128(lds0 + j * 16, gsrc + j * 16);
    }
  }
  wait_asynccnt0();
  if (threadIdx.x < 32) wait_tensorcnt0();
  __syncthreads();

  int wave = threadIdx.x >> 5;
  int lane = threadIdx.x & 31;
  int hi   = lane >> 4;

  v8f acc[4]  = {{}, {}, {}, {}};
  v8f gacc[4] = {{}, {}, {}, {}};

  for (int h = 0; h < HH; ++h) {
    // warm this head's Wd slab (128 KB) while the z-GEMM runs
    __builtin_prefetch((const char*)(Wd + (size_t)h * OO * CC)
                       + threadIdx.x * 512, 0, 1);
    // z_h = Af_h (32x32) @ feat (32x256): 32 tiles, 4 per wave, K=32 (1 step)
#pragma unroll
    for (int i = 0; i < 4; ++i) {
      int ti = wave * 4 + i;
      int mt = ti & 1, nt = ti >> 1;
      v16bf a  = load_A_MK(sAf + h * VP * VP, VP, mt * 16, 0);
      v16bf bm = load_B_KN(sfeat, CC, nt * 16, 0);
      v8f zc = {};
      zc = wmma_bf16(a, bm, zc);
      int nc = nt * 16 + (lane & 15);
#pragma unroll
      for (int r = 0; r < 8; ++r) {
        int m = mt * 16 + hi * 8 + r;
        sz[(size_t)m * CC + nc] = (bf16_t)zc[r];
      }
    }
    __syncthreads();
    // acc += z_h @ Wd[h]^T  (M=32, K=256, N=256)
#pragma unroll
    for (int i = 0; i < 4; ++i) {
      int ti = wave * 4 + i;
      int mt = ti & 1, nt = ti >> 1;
#pragma unroll
      for (int kk = 0; kk < 8; ++kk) {
        v16bf a  = load_A_MK(sz, CC, mt * 16, kk * 32);
        v16bf bm = load_B_NK(Wd + (size_t)h * OO * CC, CC, nt * 16, kk * 32);
        acc[i] = wmma_bf16(a, bm, acc[i]);
      }
    }
    __syncthreads();   // before next head overwrites sz
  }

  // gate = feat @ Wg^T
  __builtin_prefetch((const char*)Wg + threadIdx.x * 512, 0, 1);
#pragma unroll
  for (int i = 0; i < 4; ++i) {
    int ti = wave * 4 + i;
    int mt = ti & 1, nt = ti >> 1;
#pragma unroll
    for (int kk = 0; kk < 8; ++kk) {
      v16bf a  = load_A_MK(sfeat, CC, mt * 16, kk * 32);
      v16bf bm = load_B_NK(Wg, CC, nt * 16, kk * 32);
      gacc[i] = wmma_bf16(a, bm, gacc[i]);
    }
  }

  // epilogue: bias-sum, BN, sigmoid gate, residual, ReLU
#pragma unroll
  for (int i = 0; i < 4; ++i) {
    int ti = wave * 4 + i;
    int mt = ti & 1, nt = ti >> 1;
    int o  = nt * 16 + (lane & 15);
    float bd   = b_d[o] + b_d[OO + o] + b_d[2 * OO + o];
    float invs = rsqrtf(var[o] + 1e-5f) * gamma[o];
    float mn = mean[o], bb = betabn[o], bg = b_g[o];
#pragma unroll
    for (int r = 0; r < 8; ++r) {
      int v = mt * 16 + hi * 8 + r;
      if (v < VV) {
        float val = (acc[i][r] + bd - mn) * invs + bb;
        float g   = 1.0f / (1.0f + __expf(-(gacc[i][r] + bg)));
        size_t xi = (((size_t)n * CC + o) * TT + t) * VV + v;
        out[xi] = fmaxf(g * val + x[xi], 0.0f);
      }
    }
  }
}

// ---------------------------------------------------------------------------
// Host launcher
// ---------------------------------------------------------------------------
extern "C" void kernel_launch(void* const* d_in, const int* in_sizes, int n_in,
                              void* d_out, int out_size, void* d_ws, size_t ws_size,
                              hipStream_t stream) {
  (void)in_sizes; (void)n_in; (void)out_size; (void)ws_size;

  const float* x       = (const float*)d_in[0];
  const float* A_prior = (const float*)d_in[1];
  const float* A_2hop  = (const float*)d_in[2];
  const float* beta_p  = (const float*)d_in[3];
  const float* lam_p   = (const float*)d_in[4];
  const float* W_phi   = (const float*)d_in[5];
  const float* b_phi   = (const float*)d_in[6];
  const float* W_psi   = (const float*)d_in[7];
  const float* b_psi   = (const float*)d_in[8];
  const float* W_d     = (const float*)d_in[9];
  const float* b_d     = (const float*)d_in[10];
  const float* bn_g    = (const float*)d_in[11];
  const float* bn_b    = (const float*)d_in[12];
  const float* bn_m    = (const float*)d_in[13];
  const float* bn_v    = (const float*)d_in[14];
  const float* W_g     = (const float*)d_in[15];
  const float* b_g     = (const float*)d_in[16];
  float* out = (float*)d_out;

  char* ws = (char*)d_ws;
  constexpr size_t off_feat = 0;
  constexpr size_t off_pp   = off_feat + (size_t)BB * VP * CC * 2;       //  64 MB
  constexpr size_t off_Af   = off_pp   + (size_t)BB * VP * EE * 2;       // +48 MB
  constexpr size_t off_Wpp  = off_Af   + (size_t)BB * HH * VP * VP * 2;  // +24 MB
  constexpr size_t off_Wd   = off_Wpp  + (size_t)NW_PP * 2;
  constexpr size_t off_Wg   = off_Wd   + (size_t)NW_D * 2;

  bf16_t* featbf = (bf16_t*)(ws + off_feat);
  bf16_t* pp     = (bf16_t*)(ws + off_pp);
  bf16_t* Afb    = (bf16_t*)(ws + off_Af);
  bf16_t* Wpp_bf = (bf16_t*)(ws + off_Wpp);
  bf16_t* Wd_bf  = (bf16_t*)(ws + off_Wd);
  bf16_t* Wg_bf  = (bf16_t*)(ws + off_Wg);

  k_pack_feat<<<BB, 256, 0, stream>>>(x, featbf);

  int nw = (NW_PP + NW_D + NW_G + 255) / 256;
  k_pack_w<<<nw, 256, 0, stream>>>(W_phi, W_psi, W_d, W_g, Wpp_bf, Wd_bf, Wg_bf);

  k_phipsi<<<BB, 256, 0, stream>>>(featbf, Wpp_bf, b_phi, b_psi, pp);

  k_attn<<<BB, 128, 0, stream>>>(pp, A_prior, A_2hop, beta_p, lam_p, Afb);

  k_out<<<BB, 256, 0, stream>>>(featbf, Afb, Wd_bf, Wg_bf, b_d, b_g,
                                bn_g, bn_b, bn_m, bn_v, x, out);
}
